// LinformerHead_20890720928060
// MI455X (gfx1250) — matrix operand order repaired and checked by reading, not compiled, not run on hardware
//
#include <hip/hip_runtime.h>

typedef __attribute__((ext_vector_type(16))) __bf16 v16bf;
typedef __attribute__((ext_vector_type(8)))  float  v8f;

#define B_ 8
#define T_ 2048
#define C_ 1024
#define H_ 64
#define SCALE_ 0.0625f   // 1/sqrt(256)

union U32B { uint4 u[2]; v16bf v; };

static __device__ __forceinline__ v16bf pack2(uint4 lo, uint4 hi) {
  U32B t; t.u[0] = lo; t.u[1] = hi; return t.v;
}

// ---------------------------------------------------------------------------
// Kernel 0: fold E into Wk, F into Wv; convert all weights to bf16.
//   Wk'[d][c] = sum_h E[h][d] * Wk[h][c]   (so k_proj = x @ Wk'^T)
//   Wv'[d][c] = sum_h F[h][d] * Wv[h][c]
// ---------------------------------------------------------------------------
__global__ void fuse_weights(const float* __restrict__ Wq, const float* __restrict__ Wk,
                             const float* __restrict__ Wv, const float* __restrict__ E,
                             const float* __restrict__ F,
                             __bf16* __restrict__ wqb, __bf16* __restrict__ wkb,
                             __bf16* __restrict__ wvb) {
  const int idx = blockIdx.x * blockDim.x + threadIdx.x;   // 0..65535
  const int c = idx & (C_ - 1);
  const int d = idx >> 10;
  float sk = 0.f, sv = 0.f;
#pragma unroll 8
  for (int h = 0; h < H_; ++h) {
    sk += E[h * H_ + d] * Wk[h * C_ + c];
    sv += F[h * H_ + d] * Wv[h * C_ + c];
  }
  wqb[idx] = (__bf16)Wq[idx];
  wkb[idx] = (__bf16)sk;
  wvb[idx] = (__bf16)sv;
}

// ---------------------------------------------------------------------------
// Kernel 1: q / k_proj / v_proj = x @ W^T  (M=16384, K=1024, N=64), bf16 WMMA.
// grid = (M/16, 3), block = 128 (4 waves; wave w owns N-tile w).
// q is pre-scaled by SCALE_. v_proj is stored TRANSPOSED as vpT[B][H][T] so
// the attention kernel's P@V B-tiles are contiguous 128-bit loads.
// ---------------------------------------------------------------------------
__global__ __launch_bounds__(128) void proj_gemm(
    const float* __restrict__ x,
    const __bf16* __restrict__ wqb, const __bf16* __restrict__ wkb,
    const __bf16* __restrict__ wvb,
    __bf16* __restrict__ qb, __bf16* __restrict__ kpb, __bf16* __restrict__ vpT) {
  const int lane = threadIdx.x & 31;
  const int wv   = threadIdx.x >> 5;      // N-tile 0..3
  const int half = lane >> 4;
  const int l16  = lane & 15;
  const int mtile = blockIdx.x;           // 0..1023
  const int p     = blockIdx.y;           // 0:q 1:kp 2:vpT

  const __bf16* W = (p == 0) ? wqb : (p == 1) ? wkb : wvb;
  const float scale = (p == 0) ? SCALE_ : 1.0f;

  // A: 16x32 bf16 tile of x. Lane l16 = row; lane-half selects K subrange
  // (lanes 0-15: K 0-7 & 16-23 ; lanes 16-31: K 8-15 & 24-31) per ISA layout.
  const float*  xr = x + (size_t)(mtile * 16 + l16) * C_ + half * 8;
  // B: 32x16 bf16 weight tile. Lane l16 = column (output feature);
  // same K split as A (lanes 0-15: K 0-7 & 16-23 ; lanes 16-31: K 8-15 & 24-31).
  const __bf16* wr = W + (size_t)(wv * 16 + l16) * C_ + half * 8;

  v8f acc = {};
  for (int k = 0; k < C_; k += 32) {
    const float4 f0 = *(const float4*)(xr + k);
    const float4 f1 = *(const float4*)(xr + k + 4);
    const float4 g0 = *(const float4*)(xr + k + 16);
    const float4 g1 = *(const float4*)(xr + k + 20);
    v16bf a;
    a[0] = (__bf16)f0.x; a[1] = (__bf16)f0.y; a[2]  = (__bf16)f0.z; a[3]  = (__bf16)f0.w;
    a[4] = (__bf16)f1.x; a[5] = (__bf16)f1.y; a[6]  = (__bf16)f1.z; a[7]  = (__bf16)f1.w;
    a[8] = (__bf16)g0.x; a[9] = (__bf16)g0.y; a[10] = (__bf16)g0.z; a[11] = (__bf16)g0.w;
    a[12] = (__bf16)g1.x; a[13] = (__bf16)g1.y; a[14] = (__bf16)g1.z; a[15] = (__bf16)g1.w;

    const v16bf b = pack2(*(const uint4*)(wr + k), *(const uint4*)(wr + k + 16));

    acc = __builtin_amdgcn_wmma_f32_16x16x32_bf16(false, a, false, b,
                                                  (short)0, acc, false, false);
  }

  // C/D layout: lane holds N = l16, rows r + 8*half in VGPR r.
#pragma unroll
  for (int r = 0; r < 8; ++r) {
    const int row = mtile * 16 + r + half * 8;   // flat index over B*T
    if (p == 2) {
      const int b = row >> 11;                   // row / T_
      const int t = row & (T_ - 1);
      vpT[(size_t)b * (H_ * T_) + (size_t)(wv * 16 + l16) * T_ + t] =
          (__bf16)acc[r];
    } else {
      __bf16* O = (p == 0) ? qb : kpb;
      O[(size_t)row * H_ + wv * 16 + l16] = (__bf16)(acc[r] * scale);
    }
  }
}

// ---------------------------------------------------------------------------
// Kernel 2: causal flash attention. One wave per 32-query tile (two 16-row
// A tiles share every kp/vpT B-register load and P-transpose round trip);
// key loop in steps of 32 (two 16-key score tiles -> P is a 16x32 A-matrix).
// grid = (T/128, B), block = 128 (4 independent waves; no workgroup barriers —
// per-wave LDS staging synchronized with s_wait_dscnt).
// ---------------------------------------------------------------------------
__global__ __launch_bounds__(128) void flash_attn(
    const __bf16* __restrict__ qb, const __bf16* __restrict__ kpb,
    const __bf16* __restrict__ vpT, float* __restrict__ out) {
  __shared__ __bf16 Plds[4][32][32];   // per-wave P staging (C-layout -> A-layout)

  const int lane = threadIdx.x & 31;
  const int w    = threadIdx.x >> 5;
  const int half = lane >> 4;
  const int n    = lane & 15;          // doubles as A-row and C/D-column index
  const int bidx = blockIdx.y;
  const int qi   = blockIdx.x * 4 + w;
  const int t0   = qi * 32;
  const size_t bo  = (size_t)bidx * T_ * H_;   // [B,T,H] batch offset
  const size_t boT = (size_t)bidx * H_ * T_;   // [B,H,T] batch offset (vpT)

  // Hoisted Q A-regs: two M-tiles x two K-steps (K = H = 64).
  v16bf qa[2][2];
#pragma unroll
  for (int mt = 0; mt < 2; ++mt) {
    const __bf16* qr = qb + bo + (size_t)(t0 + mt * 16 + n) * H_ + half * 8;
    qa[mt][0] = pack2(*(const uint4*)(qr),      *(const uint4*)(qr + 16));
    qa[mt][1] = pack2(*(const uint4*)(qr + 32), *(const uint4*)(qr + 48));
  }

  v8f o[2][4] = {{{}, {}, {}, {}}, {{}, {}, {}, {}}};
  float mrow[2][8], lrow[2][8];
#pragma unroll
  for (int mt = 0; mt < 2; ++mt)
#pragma unroll
    for (int r = 0; r < 8; ++r) { mrow[mt][r] = -3.0e38f; lrow[mt][r] = 0.f; }

  const int nkeys = t0 + 32;
  for (int s0 = 0; s0 < nkeys; s0 += 32) {
    if (s0 + 32 < nkeys) {
      __builtin_prefetch(kpb + bo + (size_t)(s0 + 32 + n) * H_, 0, 1);
      __builtin_prefetch(vpT + boT + (size_t)(lane << 2) * T_ + s0 + 32, 0, 1);
    }

    // kp B-tiles: shared by both M-tiles. b[j][kstep].
    v16bf kb[2][2];
#pragma unroll
    for (int j = 0; j < 2; ++j) {
      const __bf16* kr = kpb + bo + (size_t)(s0 + 16 * j + n) * H_ + half * 8;
      kb[j][0] = pack2(*(const uint4*)(kr),      *(const uint4*)(kr + 16));
      kb[j][1] = pack2(*(const uint4*)(kr + 32), *(const uint4*)(kr + 48));
    }

#pragma unroll
    for (int mt = 0; mt < 2; ++mt) {
      // ---- scores: S = q @ kp^T for two 16-key tiles ------------------
      v8f s[2];
#pragma unroll
      for (int j = 0; j < 2; ++j) {
        v8f sj = {};
        sj = __builtin_amdgcn_wmma_f32_16x16x32_bf16(false, qa[mt][0], false, kb[j][0],
                                                     (short)0, sj, false, false);
        sj = __builtin_amdgcn_wmma_f32_16x16x32_bf16(false, qa[mt][1], false, kb[j][1],
                                                     (short)0, sj, false, false);
        s[j] = sj;
      }

      // ---- causal mask + online softmax (16-lane shuffle reductions) --
#pragma unroll
      for (int r = 0; r < 8; ++r) {
        const int trow = t0 + mt * 16 + r + half * 8;
        float v0 = (s0 + n      <= trow) ? s[0][r] : -3.0e38f;
        float v1 = (s0 + 16 + n <= trow) ? s[1][r] : -3.0e38f;
        float mx = fmaxf(v0, v1);
        mx = fmaxf(mx, __shfl_xor(mx, 1, 32));
        mx = fmaxf(mx, __shfl_xor(mx, 2, 32));
        mx = fmaxf(mx, __shfl_xor(mx, 4, 32));
        mx = fmaxf(mx, __shfl_xor(mx, 8, 32));
        const float mnew = fmaxf(mrow[mt][r], mx);
        const float corr = __expf(mrow[mt][r] - mnew);
        const float p0 = __expf(v0 - mnew);
        const float p1 = __expf(v1 - mnew);
        float rs = p0 + p1;
        rs += __shfl_xor(rs, 1, 32);
        rs += __shfl_xor(rs, 2, 32);
        rs += __shfl_xor(rs, 4, 32);
        rs += __shfl_xor(rs, 8, 32);
        lrow[mt][r] = lrow[mt][r] * corr + rs;
        mrow[mt][r] = mnew;
        o[mt][0][r] *= corr; o[mt][1][r] *= corr;
        o[mt][2][r] *= corr; o[mt][3][r] *= corr;
        Plds[w][mt * 16 + r + half * 8][n]      = (__bf16)p0;
        Plds[w][mt * 16 + r + half * 8][16 + n] = (__bf16)p1;
      }
    }

    // Intra-wave LDS RAW: drain DS counter (no workgroup barrier — waves
    // have different causal trip counts).
    asm volatile("s_wait_dscnt 0" ::: "memory");

    // P (2 x 16x32) re-read in WMMA A layout.
    v16bf pa[2];
#pragma unroll
    for (int mt = 0; mt < 2; ++mt)
      pa[mt] = pack2(*(const uint4*)&Plds[w][mt * 16 + n][half * 8],
                     *(const uint4*)&Plds[w][mt * 16 + n][half * 8 + 16]);

    // ---- O += P @ v_proj: vpT gives contiguous key-major B tiles ------
#pragma unroll
    for (int nt = 0; nt < 4; ++nt) {
      const __bf16* vr = vpT + boT + (size_t)(nt * 16 + n) * T_ + s0 + half * 8;
      const v16bf bb = pack2(*(const uint4*)(vr), *(const uint4*)(vr + 16));
      o[0][nt] = __builtin_amdgcn_wmma_f32_16x16x32_bf16(false, pa[0], false, bb,
                                                         (short)0, o[0][nt], false, false);
      o[1][nt] = __builtin_amdgcn_wmma_f32_16x16x32_bf16(false, pa[1], false, bb,
                                                         (short)0, o[1][nt], false, false);
    }
  }

  // ---- normalize and store (f32 output) -------------------------------
#pragma unroll
  for (int mt = 0; mt < 2; ++mt) {
#pragma unroll
    for (int r = 0; r < 8; ++r) {
      const float inv = 1.0f / lrow[mt][r];
      const size_t rowb = bo + (size_t)(t0 + mt * 16 + r + half * 8) * H_;
      out[rowb + 0  + n] = o[mt][0][r] * inv;
      out[rowb + 16 + n] = o[mt][1][r] * inv;
      out[rowb + 32 + n] = o[mt][2][r] * inv;
      out[rowb + 48 + n] = o[mt][3][r] * inv;
    }
  }
}

// ---------------------------------------------------------------------------
extern "C" void kernel_launch(void* const* d_in, const int* in_sizes, int n_in,
                              void* d_out, int out_size, void* d_ws, size_t ws_size,
                              hipStream_t stream) {
  const float* x  = (const float*)d_in[0];
  const float* Wq = (const float*)d_in[1];
  const float* Wk = (const float*)d_in[2];
  const float* Wv = (const float*)d_in[3];
  const float* E  = (const float*)d_in[4];
  const float* F  = (const float*)d_in[5];
  float* out = (float*)d_out;

  char* ws = (char*)d_ws;
  __bf16* wqb = (__bf16*)(ws);                                   // 128 KB
  __bf16* wkb = (__bf16*)(ws + (128 << 10));                     // 128 KB
  __bf16* wvb = (__bf16*)(ws + (256 << 10));                     // 128 KB
  __bf16* qb  = (__bf16*)(ws + (512 << 10));                     // 2 MB
  __bf16* kpb = (__bf16*)(ws + (512 << 10) + (2 << 20));         // 2 MB
  __bf16* vpT = (__bf16*)(ws + (512 << 10) + (4 << 20));         // 2 MB, [B,H,T]

  fuse_weights<<<dim3(H_ * C_ / 256), dim3(256), 0, stream>>>(Wq, Wk, Wv, E, F,
                                                              wqb, wkb, wvb);
  proj_gemm<<<dim3(B_ * T_ / 16, 3), dim3(128), 0, stream>>>(x, wqb, wkb, wvb,
                                                             qb, kpb, vpT);
  flash_attn<<<dim3(T_ / 128, B_), dim3(128), 0, stream>>>(qb, kpb, vpT, out);
}
